// EnhancedSupplyGNN_81853486727231
// MI455X (gfx1250) — matrix operand order repaired and checked by reading, compiled, not run on hardware
//
#include <hip/hip_runtime.h>
#include <hip/hip_bf16.h>

#define N_NODES 4096
#define E_EDGES 262144
#define F_NODE 32
#define F_EDGE 8
#define HID 128
#define HEADS 4
#define CDIM 32
#define EDIM 32
#define DOUT 64
#define AH 4
#define AHD 16
#define E2 (E_EDGES + N_NODES)
#define CHUNK 33280               /* E2 / 8, multiple of 16 */
#define INV_STD 0.9999950000374996f /* 1/sqrt(1+1e-5) */

typedef __attribute__((ext_vector_type(16))) _Float16 v16h;
typedef __attribute__((ext_vector_type(8)))  float    v8f;

enum { ACT_NONE = 0, ACT_RELU = 1, ACT_SIGMOID = 2 };

// ---------- float ordered-uint helpers for atomic max on floats ----------
__device__ __forceinline__ unsigned f2ord(float f) {
  unsigned u = __float_as_uint(f);
  return (u & 0x80000000u) ? ~u : (u | 0x80000000u);
}
__device__ __forceinline__ float ord2f(unsigned u) {
  u = (u & 0x80000000u) ? (u & 0x7FFFFFFFu) : ~u;
  return __uint_as_float(u);
}
__device__ __forceinline__ void atomAddF(float* p, float v) {
  unsafeAtomicAdd(p, v);   // lowers to global_atomic_add_f32 on gfx1250
}
__device__ __forceinline__ float applyAct(float v, int act) {
  if (act == ACT_RELU)    return fmaxf(v, 0.0f);
  if (act == ACT_SIGMOID) return 1.0f / (1.0f + __expf(-v));
  return v;
}

// =====================================================================
// Pack B[K x Nc] (row-major f32) into WMMA-native f16 layout:
//   pB[((tn*nKb + kb)*32 + lane)*16 + e]
//     = B[kb*32 + (lane/16)*8 + (e%8) + (e/8)*16][tn*16 + lane%16]   (0 if OOB)
// Each lane's 16 B values become one contiguous 32-byte chunk;
// K/Nc raggedness is zero-padded here so GEMMs need no guards.
// =====================================================================
__global__ void pack_b_kernel(const float* __restrict__ B,
                              _Float16* __restrict__ pB,
                              int K, int Nc, int total) {
  int idx = blockIdx.x * blockDim.x + threadIdx.x;
  if (idx >= total) return;
  const int e    = idx & 15;
  const int lane = (idx >> 4) & 31;
  const int blk  = idx >> 9;              // tn*nKb + kb
  const int nKb  = (K + 31) >> 5;
  const int kb   = blk % nKb;
  const int tn   = blk / nKb;
  const int col  = tn * 16 + (lane & 15);
  const int kk   = kb * 32 + ((lane >> 4) << 3) + (e & 7) + ((e >> 3) << 4);
  float v = (kk < K && col < Nc) ? B[(long long)kk * Nc + col] : 0.0f;
  pB[idx] = (_Float16)v;
}

// =====================================================================
// WMMA GEMM, 4 N-tiles per wave (16x64 strip), packed-B.
// Requires M%16==0, K%32==0, Nc%64==0. One A fetch -> 4 WMMAs.
// =====================================================================
__global__ void wmma_gemm_nt4_pk_kernel(const float* __restrict__ A,
                                        const _Float16* __restrict__ pB,
                                        const float* __restrict__ bias,
                                        float* __restrict__ C,
                                        int M, int K, int Nc, int act) {
  const int lane = threadIdx.x & 31;
  const long long wave = ((long long)blockIdx.x * blockDim.x + threadIdx.x) >> 5;
  const int stripsN = Nc >> 6;
  const int tilesM  = M >> 4;
  if (wave >= (long long)tilesM * stripsN) return;   // wave-uniform exit
  const int sN = (int)(wave % stripsN);
  const int tM = (int)(wave / stripsN);
  const int m  = lane & 15;
  const int hi = lane >> 4;
  const int row  = tM * 16 + m;
  const int col0 = sN * 64 + m;
  const int nKb  = K >> 5;

  const float* __restrict__ Arow = A + (long long)row * K;
  const long long tileStride = (long long)nKb * 512;          // f16 per N-tile
  const _Float16* __restrict__ pb0 =
      pB + (long long)(sN * 4) * tileStride + (long long)lane * 16;

  v8f acc0 = {}, acc1 = {}, acc2 = {}, acc3 = {};
  for (int kb = 0; kb < nKb; ++kb) {
    v16h a;
    const int base0 = kb * 32 + (hi << 3);
#pragma unroll
    for (int e = 0; e < 8; ++e) {
      a[e]     = (_Float16)Arow[base0 + e];
      a[8 + e] = (_Float16)Arow[base0 + 16 + e];
    }
    const _Float16* __restrict__ pk = pb0 + (long long)kb * 512;
    v16h b0 = *(const v16h*)(pk);
    v16h b1 = *(const v16h*)(pk + tileStride);
    v16h b2 = *(const v16h*)(pk + 2 * tileStride);
    v16h b3 = *(const v16h*)(pk + 3 * tileStride);
    acc0 = __builtin_amdgcn_wmma_f32_16x16x32_f16(false, a, false, b0, (short)0, acc0, false, false);
    acc1 = __builtin_amdgcn_wmma_f32_16x16x32_f16(false, a, false, b1, (short)0, acc1, false, false);
    acc2 = __builtin_amdgcn_wmma_f32_16x16x32_f16(false, a, false, b2, (short)0, acc2, false, false);
    acc3 = __builtin_amdgcn_wmma_f32_16x16x32_f16(false, a, false, b3, (short)0, acc3, false, false);
  }

  const float bv0 = bias ? bias[col0]      : 0.0f;
  const float bv1 = bias ? bias[col0 + 16] : 0.0f;
  const float bv2 = bias ? bias[col0 + 32] : 0.0f;
  const float bv3 = bias ? bias[col0 + 48] : 0.0f;
#pragma unroll
  for (int r = 0; r < 8; ++r) {
    const int orow = tM * 16 + r + (hi << 3);
    float* __restrict__ Cp = C + (long long)orow * Nc + col0;
    Cp[0]  = applyAct(acc0[r] + bv0, act);
    Cp[16] = applyAct(acc1[r] + bv1, act);
    Cp[32] = applyAct(acc2[r] + bv2, act);
    Cp[48] = applyAct(acc3[r] + bv3, act);
  }
}

// =====================================================================
// Generic WMMA GEMM (single 16x16 tile per wave), packed-B.
// M%16==0 required. Ragged K: clamped/masked A tail (B pre-padded).
// Ragged Nc (template): store guard only.
// =====================================================================
template <bool NCR>
__global__ void wmma_gemm_pk_kernel(const float* __restrict__ A,
                                    const _Float16* __restrict__ pB,
                                    const float* __restrict__ bias,
                                    float* __restrict__ C,
                                    int M, int K, int Nc, int act) {
  const int lane = threadIdx.x & 31;
  const long long wave = ((long long)blockIdx.x * blockDim.x + threadIdx.x) >> 5;
  const int tilesN = (Nc + 15) >> 4;
  const int tilesM = M >> 4;
  if (wave >= (long long)tilesM * tilesN) return;   // wave-uniform exit
  const int tN = (int)(wave % tilesN);
  const int tM = (int)(wave / tilesN);
  const int m  = lane & 15;
  const int hi = lane >> 4;
  const int row = tM * 16 + m;
  const int col = tN * 16 + m;
  const int nKb   = (K + 31) >> 5;
  const int Kfull = K >> 5;

  const float* __restrict__ Arow = A + (long long)row * K;
  const _Float16* __restrict__ pb0 =
      pB + (long long)tN * nKb * 512 + (long long)lane * 16;

  v8f acc = {};
  for (int kb = 0; kb < Kfull; ++kb) {
    v16h a;
    const int base0 = kb * 32 + (hi << 3);
#pragma unroll
    for (int e = 0; e < 8; ++e) {
      a[e]     = (_Float16)Arow[base0 + e];
      a[8 + e] = (_Float16)Arow[base0 + 16 + e];
    }
    v16h b = *(const v16h*)(pb0 + (long long)kb * 512);
    acc = __builtin_amdgcn_wmma_f32_16x16x32_f16(false, a, false, b, (short)0, acc, false, false);
  }
  if (Kfull < nKb) {      // ragged-K tail: clamp addresses, mask values
    v16h a;
    const int base0 = Kfull * 32 + (hi << 3);
    const int base1 = base0 + 16;
#pragma unroll
    for (int e = 0; e < 8; ++e) {
      int ka0 = base0 + e, ka1 = base1 + e;
      float v0 = Arow[ka0 < K ? ka0 : K - 1];
      float v1 = Arow[ka1 < K ? ka1 : K - 1];
      a[e]     = (_Float16)(ka0 < K ? v0 : 0.0f);
      a[8 + e] = (_Float16)(ka1 < K ? v1 : 0.0f);
    }
    v16h b = *(const v16h*)(pb0 + (long long)Kfull * 512);
    acc = __builtin_amdgcn_wmma_f32_16x16x32_f16(false, a, false, b, (short)0, acc, false, false);
  }

  if (!NCR || col < Nc) {
    const float bv = bias ? bias[col] : 0.0f;
#pragma unroll
    for (int r = 0; r < 8; ++r) {
      const int orow = tM * 16 + r + (hi << 3);
      C[(long long)orow * Nc + col] = applyAct(acc[r] + bv, act);
    }
  }
}

// ---------------------------------------------------------------------
static inline void pack_b(const float* B, _Float16* pB, int K, int Nc,
                          hipStream_t s) {
  const int nKb = (K + 31) >> 5;
  const int tilesN = (Nc + 15) >> 4;
  const int total = tilesN * nKb * 512;
  pack_b_kernel<<<(total + 255) / 256, 256, 0, s>>>(B, pB, K, Nc, total);
}

static inline void launch_gemm_packed(const float* A, const _Float16* pB,
                                      const float* bias, float* C,
                                      int M, int K, int Nc, int act,
                                      hipStream_t s) {
  const int tpb = 256;
  if ((Nc & 63) == 0 && (K & 31) == 0) {
    const long long waves = (long long)(M >> 4) * (Nc >> 6);
    const int blocks = (int)((waves * 32 + tpb - 1) / tpb);
    wmma_gemm_nt4_pk_kernel<<<blocks, tpb, 0, s>>>(A, pB, bias, C, M, K, Nc, act);
  } else {
    const long long waves = (long long)(M >> 4) * ((Nc + 15) >> 4);
    const int blocks = (int)((waves * 32 + tpb - 1) / tpb);
    if ((Nc & 15) == 0)
      wmma_gemm_pk_kernel<false><<<blocks, tpb, 0, s>>>(A, pB, bias, C, M, K, Nc, act);
    else
      wmma_gemm_pk_kernel<true><<<blocks, tpb, 0, s>>>(A, pB, bias, C, M, K, Nc, act);
  }
}

static inline void launch_gemm(const float* A, const float* B, const float* bias,
                               float* C, int M, int K, int Nc, int act,
                               _Float16* pB, hipStream_t s) {
  pack_b(B, pB, K, Nc, s);
  launch_gemm_packed(A, pB, bias, C, M, K, Nc, act, s);
}

// =====================================================================
// Self-loop edge attr: deg + segment_sum(ea, dst), then divide
// =====================================================================
__global__ void loop_sum_kernel(const int* __restrict__ dstA,
                                const float* __restrict__ ea,
                                float* __restrict__ loopRegion,
                                float* __restrict__ deg) {
  int e = blockIdx.x * blockDim.x + threadIdx.x;
  if (e >= E_EDGES) return;
  int d = dstA[e];
  atomAddF(&deg[d], 1.0f);
  const float* r = ea + (long long)e * EDIM;
  float* o = loopRegion + (long long)d * EDIM;
#pragma unroll
  for (int j = 0; j < EDIM; ++j) atomAddF(&o[j], r[j]);
}

__global__ void loop_div_kernel(float* __restrict__ loopRegion,
                                const float* __restrict__ deg) {
  int idx = blockIdx.x * blockDim.x + threadIdx.x;
  if (idx >= N_NODES * EDIM) return;
  loopRegion[idx] /= fmaxf(deg[idx / EDIM], 1.0f);
}

// =====================================================================
// GATv2 attention pieces
// =====================================================================
__global__ void gat_logit_kernel(const float* __restrict__ hp,
                                 const float* __restrict__ ep,   // chunk [nE x HID]
                                 const float* __restrict__ att,  // [HEADS*CDIM]
                                 const int* __restrict__ srcA,
                                 const int* __restrict__ dstA,
                                 float* __restrict__ logit,      // [E2*HEADS]
                                 unsigned* __restrict__ mx_u,    // [N*HEADS]
                                 int e0, int nE) {
  int idx = blockIdx.x * blockDim.x + threadIdx.x;
  if (idx >= nE * HEADS) return;
  const int el = idx >> 2, h = idx & 3;
  const int e = e0 + el;
  const int s = (e < E_EDGES) ? srcA[e] : (e - E_EDGES);
  const int d = (e < E_EDGES) ? dstA[e] : (e - E_EDGES);
  const float* hs = hp + (long long)s * HID + h * CDIM;
  const float* hd = hp + (long long)d * HID + h * CDIM;
  const float* ev = ep + (long long)el * HID + h * CDIM;
  const float* av = att + h * CDIM;
  float lg = 0.0f;
#pragma unroll
  for (int c = 0; c < CDIM; ++c) {
    float t = hs[c] + hd[c] + ev[c];
    t = (t > 0.0f) ? t : 0.2f * t;           // leaky_relu(0.2)
    lg += t * av[c];
  }
  logit[(long long)e * HEADS + h] = lg;
  atomicMax(&mx_u[d * HEADS + h], f2ord(lg));
}

__global__ void gat_exp_kernel(const int* __restrict__ dstA,
                               float* __restrict__ logit,
                               const unsigned* __restrict__ mx_u,
                               float* __restrict__ den) {
  long long idx = (long long)blockIdx.x * blockDim.x + threadIdx.x;
  if (idx >= (long long)E2 * HEADS) return;
  const int e = (int)(idx >> 2), h = (int)(idx & 3);
  const int d = (e < E_EDGES) ? dstA[e] : (e - E_EDGES);
  float ex = __expf(logit[idx] - ord2f(mx_u[d * HEADS + h]));
  logit[idx] = ex;
  atomAddF(&den[d * HEADS + h], ex);
}

__global__ void gat_agg_kernel(const int* __restrict__ srcA,
                               const int* __restrict__ dstA,
                               const float* __restrict__ logit,
                               const float* __restrict__ den,
                               const float* __restrict__ hp,
                               float* __restrict__ agg) {
  long long idx = (long long)blockIdx.x * blockDim.x + threadIdx.x;
  if (idx >= (long long)E2 * HEADS) return;
  const int e = (int)(idx >> 2), h = (int)(idx & 3);
  const int s = (e < E_EDGES) ? srcA[e] : (e - E_EDGES);
  const int d = (e < E_EDGES) ? dstA[e] : (e - E_EDGES);
  const float alpha = logit[idx] / (den[d * HEADS + h] + 1e-16f);
  const float* hs = hp + (long long)s * HID + h * CDIM;
  float* o = agg + (long long)d * HID + h * CDIM;
#pragma unroll
  for (int c = 0; c < CDIM; ++c) atomAddF(&o[c], alpha * hs[c]);
}

__global__ void gat_post_kernel(float* __restrict__ hbuf,
                                const float* __restrict__ agg,
                                const float* __restrict__ bo,
                                const float* __restrict__ g,
                                const float* __restrict__ b) {
  int idx = blockIdx.x * blockDim.x + threadIdx.x;
  if (idx >= N_NODES * HID) return;
  const int j = idx & (HID - 1);
  float v = agg[idx] + bo[j];
  v = g[j] * (v * INV_STD) + b[j];           // BN eval mode
  v = fmaxf(v, 0.0f);                        // relu
  hbuf[idx] += v;                            // residual
}

// node-type select: proc[n,:] = z[type(n)][n,:]
__global__ void select_kernel(const float* __restrict__ zbuf,
                              const int* __restrict__ nt,
                              float* __restrict__ proc) {
  int idx = blockIdx.x * blockDim.x + threadIdx.x;
  if (idx >= N_NODES * HID) return;
  const int n = idx >> 7;
  const int t = nt[n] & 3;
  proc[idx] = zbuf[(long long)t * N_NODES * HID + idx];
}

// =====================================================================
// Full self-attention, online softmax. One wave32 per (query, head);
// lanes stride over 4096 keys, butterfly-merge (max,sum,acc) at end.
// =====================================================================
__global__ void attn_kernel(const float* __restrict__ q,
                            const float* __restrict__ k,
                            const float* __restrict__ v,
                            float* __restrict__ out) {
  const int lane = threadIdx.x & 31;
  const int wave = (blockIdx.x * blockDim.x + threadIdx.x) >> 5;
  if (wave >= N_NODES * AH) return;
  const int n = wave >> 2, h = wave & 3;
  float qv[AHD];
#pragma unroll
  for (int d = 0; d < AHD; ++d) qv[d] = q[(long long)n * DOUT + h * AHD + d];
  float rmax = -1e30f, rsum = 0.0f, accv[AHD];
#pragma unroll
  for (int d = 0; d < AHD; ++d) accv[d] = 0.0f;

  for (int m = lane; m < N_NODES; m += 32) {
    const float* kr = k + (long long)m * DOUT + h * AHD;
    float s = 0.0f;
#pragma unroll
    for (int d = 0; d < AHD; ++d) s += qv[d] * kr[d];
    s *= 0.25f;                                // 1/sqrt(16)
    const float nm = fmaxf(rmax, s);
    const float corr = __expf(rmax - nm);
    const float ev = __expf(s - nm);
    rsum = rsum * corr + ev;
    const float* vr = v + (long long)m * DOUT + h * AHD;
#pragma unroll
    for (int d = 0; d < AHD; ++d) accv[d] = accv[d] * corr + ev * vr[d];
    rmax = nm;
  }
#pragma unroll
  for (int off = 16; off >= 1; off >>= 1) {
    const float om = __shfl_xor(rmax, off, 32);
    const float os = __shfl_xor(rsum, off, 32);
    const float nm = fmaxf(rmax, om);
    const float c1 = __expf(rmax - nm), c2 = __expf(om - nm);
    rsum = rsum * c1 + os * c2;
#pragma unroll
    for (int d = 0; d < AHD; ++d) {
      const float ov = __shfl_xor(accv[d], off, 32);
      accv[d] = accv[d] * c1 + ov * c2;
    }
    rmax = nm;
  }
  if (lane == 0) {
    const float inv = 1.0f / rsum;
#pragma unroll
    for (int d = 0; d < AHD; ++d)
      out[(long long)n * DOUT + h * AHD + d] = accv[d] * inv;
  }
}

// =====================================================================
// Host orchestration
// =====================================================================
extern "C" void kernel_launch(void* const* d_in, const int* in_sizes, int n_in,
                              void* d_out, int out_size, void* d_ws,
                              size_t ws_size, hipStream_t stream) {
  (void)in_sizes; (void)n_in; (void)out_size; (void)ws_size;

  const float* x         = (const float*)d_in[0];
  const float* edge_attr = (const float*)d_in[1];
  const int*   edge_idx  = (const int*)d_in[2];
  const int*   node_type = (const int*)d_in[3];
  const float* Win  = (const float*)d_in[4];
  const float* bin_ = (const float*)d_in[5];
  const float* Wedge = (const float*)d_in[6];
  const float* bedge = (const float*)d_in[7];
  const float* Wconv = (const float*)d_in[8];
  const float* bconv = (const float*)d_in[9];
  const float* Weconv = (const float*)d_in[10];
  const float* attconv = (const float*)d_in[11];
  const float* boconv = (const float*)d_in[12];
  const float* gamma = (const float*)d_in[13];
  const float* beta  = (const float*)d_in[14];
  const float* Wt1 = (const float*)d_in[15];
  const float* bt1 = (const float*)d_in[16];
  const float* Wt2 = (const float*)d_in[17];
  const float* bt2 = (const float*)d_in[18];
  const float* Wf = (const float*)d_in[19];
  const float* bf = (const float*)d_in[20];
  const float* Wqkv = (const float*)d_in[21];
  const float* bqkv = (const float*)d_in[22];
  const float* Wo = (const float*)d_in[23];
  const float* bo_attn = (const float*)d_in[24];
  const float* Wh1 = (const float*)d_in[25];
  const float* bh1 = (const float*)d_in[26];
  const float* Wh2 = (const float*)d_in[27];
  const float* bh2 = (const float*)d_in[28];
  const float* Wh3 = (const float*)d_in[29];
  const float* bh3 = (const float*)d_in[30];

  const int* srcA = edge_idx;
  const int* dstA = edge_idx + E_EDGES;

  // ---- workspace layout (all sizes multiples of 16 floats) ----
  float* ws = (float*)d_ws;
  size_t off = 0;
  auto alloc = [&](size_t n) { float* p = ws + off; off += n; return p; };
  float*    hbuf  = alloc((size_t)N_NODES * HID);
  float*    hp    = alloc((size_t)N_NODES * HID);
  float*    ea2   = alloc((size_t)E2 * EDIM);
  float*    deg   = alloc(N_NODES);
  float*    logit = alloc((size_t)E2 * HEADS);
  unsigned* mx_u  = (unsigned*)alloc((size_t)N_NODES * HEADS);
  float*    den   = alloc((size_t)N_NODES * HEADS);
  float*    agg   = alloc((size_t)N_NODES * HID);
  float*    epchk = alloc((size_t)CHUNK * HID);
  float*    zbuf  = alloc((size_t)4 * N_NODES * HID);
  float*    tmp1  = alloc((size_t)N_NODES * HID);
  float*    proc  = alloc((size_t)N_NODES * HID);
  float*    ne    = alloc((size_t)N_NODES * DOUT);
  float*    qb    = alloc((size_t)N_NODES * DOUT);
  float*    kb    = alloc((size_t)N_NODES * DOUT);
  float*    vb    = alloc((size_t)N_NODES * DOUT);
  float*    attno = alloc((size_t)N_NODES * DOUT);
  float*    h1b   = alloc((size_t)N_NODES * 32);
  float*    h2b   = alloc((size_t)N_NODES * 16);
  _Float16* pBw   = (_Float16*)alloc(16384);   // 32K f16 packed-B staging

  const int TPB = 256;

  // 1) input projection  h = relu(x @ Win + bin)
  launch_gemm(x, Win, bin_, hbuf, N_NODES, F_NODE, HID, ACT_RELU, pBw, stream);
  // 2) edge projection into ea2[0:E]
  launch_gemm(edge_attr, Wedge, bedge, ea2, E_EDGES, F_EDGE, EDIM, ACT_NONE, pBw, stream);
  // 3) self-loop attrs = mean of incoming edge attrs
  (void)hipMemsetAsync(deg, 0, N_NODES * sizeof(float), stream);
  (void)hipMemsetAsync(ea2 + (size_t)E_EDGES * EDIM, 0,
                       (size_t)N_NODES * EDIM * sizeof(float), stream);
  loop_sum_kernel<<<(E_EDGES + TPB - 1) / TPB, TPB, 0, stream>>>(
      dstA, ea2, ea2 + (size_t)E_EDGES * EDIM, deg);
  loop_div_kernel<<<(N_NODES * EDIM + TPB - 1) / TPB, TPB, 0, stream>>>(
      ea2 + (size_t)E_EDGES * EDIM, deg);

  // 4) three GATv2 layers
  for (int l = 0; l < 3; ++l) {
    const float* Wl   = Wconv + (size_t)l * HID * HID;
    const float* bl   = bconv + (size_t)l * HID;
    const float* Wel  = Weconv + (size_t)l * EDIM * HID;
    const float* attl = attconv + (size_t)l * HEADS * CDIM;
    const float* bol  = boconv + (size_t)l * HID;
    const float* gl   = gamma + (size_t)l * HID;
    const float* btl  = beta + (size_t)l * HID;

    launch_gemm(hbuf, Wl, bl, hp, N_NODES, HID, HID, ACT_NONE, pBw, stream);
    (void)hipMemsetAsync(mx_u, 0, (size_t)N_NODES * HEADS * sizeof(unsigned), stream);
    (void)hipMemsetAsync(den, 0, (size_t)N_NODES * HEADS * sizeof(float), stream);
    (void)hipMemsetAsync(agg, 0, (size_t)N_NODES * HID * sizeof(float), stream);

    pack_b(Wel, pBw, EDIM, HID, stream);   // pack Weconv[l] once per layer
    for (int e0 = 0; e0 < E2; e0 += CHUNK) {
      launch_gemm_packed(ea2 + (size_t)e0 * EDIM, pBw, nullptr, epchk,
                         CHUNK, EDIM, HID, ACT_NONE, stream);
      gat_logit_kernel<<<(CHUNK * HEADS + TPB - 1) / TPB, TPB, 0, stream>>>(
          hp, epchk, attl, srcA, dstA, logit, mx_u, e0, CHUNK);
    }
    const long long tot = (long long)E2 * HEADS;
    gat_exp_kernel<<<(int)((tot + TPB - 1) / TPB), TPB, 0, stream>>>(
        dstA, logit, mx_u, den);
    gat_agg_kernel<<<(int)((tot + TPB - 1) / TPB), TPB, 0, stream>>>(
        srcA, dstA, logit, den, hp, agg);
    gat_post_kernel<<<(N_NODES * HID + TPB - 1) / TPB, TPB, 0, stream>>>(
        hbuf, agg, bol, gl, btl);
  }

  // 5) node-type processor MLPs (compute all 4, then select)
  for (int t = 0; t < 4; ++t) {
    launch_gemm(hbuf, Wt1 + (size_t)t * HID * HID, bt1 + (size_t)t * HID,
                tmp1, N_NODES, HID, HID, ACT_RELU, pBw, stream);
    launch_gemm(tmp1, Wt2 + (size_t)t * HID * HID, bt2 + (size_t)t * HID,
                zbuf + (size_t)t * N_NODES * HID, N_NODES, HID, HID, ACT_NONE,
                pBw, stream);
  }
  select_kernel<<<(N_NODES * HID + TPB - 1) / TPB, TPB, 0, stream>>>(
      zbuf, node_type, proc);

  // 6) final projection + QKV
  launch_gemm(proc, Wf, bf, ne, N_NODES, HID, DOUT, ACT_NONE, pBw, stream);
  launch_gemm(ne, Wqkv,                   bqkv,            qb, N_NODES, DOUT, DOUT, ACT_NONE, pBw, stream);
  launch_gemm(ne, Wqkv + DOUT * DOUT,     bqkv + DOUT,     kb, N_NODES, DOUT, DOUT, ACT_NONE, pBw, stream);
  launch_gemm(ne, Wqkv + 2 * DOUT * DOUT, bqkv + 2 * DOUT, vb, N_NODES, DOUT, DOUT, ACT_NONE, pBw, stream);

  // 7) self-attention (online softmax, one wave per (query, head))
  attn_kernel<<<(N_NODES * AH * 32 + TPB - 1) / TPB, TPB, 0, stream>>>(
      qb, kb, vb, attno);

  // 8) enh = attn_out @ Wo + bo  (directly into d_out)
  float* outF = (float*)d_out;
  launch_gemm(attno, Wo, bo_attn, outF, N_NODES, DOUT, DOUT, ACT_NONE, pBw, stream);

  // 9) three prediction heads -> sigmoid outputs after enh
  float* preds = outF + (size_t)N_NODES * DOUT;
  for (int t = 0; t < 3; ++t) {
    launch_gemm(outF, Wh1 + (size_t)t * DOUT * 32, bh1 + (size_t)t * 32,
                h1b, N_NODES, DOUT, 32, ACT_RELU, pBw, stream);
    launch_gemm(h1b, Wh2 + (size_t)t * 32 * 16, bh2 + (size_t)t * 16,
                h2b, N_NODES, 32, 16, ACT_RELU, pBw, stream);
    launch_gemm(h2b, Wh3 + (size_t)t * 16 * 1, bh3 + t,
                preds + (size_t)t * N_NODES, N_NODES, 16, 1, ACT_SIGMOID,
                pBw, stream);
  }
}
